// Transformer_36103495090531
// MI455X (gfx1250) — matrix operand order repaired and checked by reading, not compiled, not run on hardware
//
#include <hip/hip_runtime.h>
#include <cstdint>

// ---------------- problem constants ----------------
#define DEPTH   6
#define E_DIM   1024
#define H_DIM   1024
#define HEADS   16
#define HH_DIM  64
#define L_SEQ   1024
#define N_B     2
#define NL      (N_B * L_SEQ)                 // 2048 token rows
#define NLH     ((size_t)NL * (size_t)H_DIM)  // 2M activation elements
#define EH      ((size_t)E_DIM * (size_t)H_DIM)
#define INV_SQRT_H 0.03125f                   // 1/sqrt(1024)

typedef __bf16 bf16_t;
typedef __attribute__((ext_vector_type(16))) __bf16 v16bf;
typedef __attribute__((ext_vector_type(8)))  __bf16 v8bf;
typedef __attribute__((ext_vector_type(8)))  float  v8f;
typedef __attribute__((ext_vector_type(4)))  int    v4i;

union Frag16 { v16bf v; v8bf h[2]; };

// ---------------- CDNA5 async memory->LDS staging ----------------
#if defined(__has_builtin)
#  if __has_builtin(__builtin_amdgcn_global_load_async_to_lds_b128)
#    define HAVE_ASYNC_LDS 1
#  endif
#endif
#ifndef HAVE_ASYNC_LDS
#  define HAVE_ASYNC_LDS 0
#endif

// 16-byte global -> LDS copy.  Async path uses GLOBAL_LOAD_ASYNC_TO_LDS_B128
// (no VGPR data movement, tracked by ASYNCcnt); fallback is load+ds_store.
// Builtin signature (probe-confirmed): (int4 global*, int4 local*, imm, imm).
__device__ __forceinline__ void stage16(void* lds_dst, const void* gsrc) {
#if HAVE_ASYNC_LDS
  auto gp = (__attribute__((address_space(1))) v4i*)(uintptr_t)gsrc;
  auto lp = (__attribute__((address_space(3))) v4i*)(unsigned)(uintptr_t)lds_dst;
  __builtin_amdgcn_global_load_async_to_lds_b128(gp, lp, 0, 0);
#else
  *(v8bf*)lds_dst = *(const v8bf*)gsrc;
#endif
}

__device__ __forceinline__ void wait_stage() {
#if HAVE_ASYNC_LDS
#  if __has_builtin(__builtin_amdgcn_s_wait_asynccnt)
  __builtin_amdgcn_s_wait_asynccnt(0);
#  else
  asm volatile("s_wait_asynccnt 0x0" ::: "memory");
#  endif
#endif
}

// ---------------- embedding: x = tok_emb[X] + pos_emb ----------------
__global__ __launch_bounds__(256) void embed_kernel(
    const int* __restrict__ X, const float* __restrict__ tok,
    const float* __restrict__ pos, float* __restrict__ xF, bf16_t* __restrict__ xB)
{
  const int nl = blockIdx.x;               // one block per (n,l) row
  const int l  = nl & (L_SEQ - 1);
  const int tk = X[nl];
  const int c  = threadIdx.x * 4;
  const float4 tv = *(const float4*)(tok + (size_t)tk * E_DIM + c);
  const float4 pv = *(const float4*)(pos + (size_t)l  * E_DIM + c);
  const float v0 = tv.x + pv.x, v1 = tv.y + pv.y, v2 = tv.z + pv.z, v3 = tv.w + pv.w;
  const size_t o = (size_t)nl * E_DIM + c;
  xF[o+0]=v0; xF[o+1]=v1; xF[o+2]=v2; xF[o+3]=v3;
  xB[o+0]=(bf16_t)v0; xB[o+1]=(bf16_t)v1; xB[o+2]=(bf16_t)v2; xB[o+3]=(bf16_t)v3;
}

// ---------------- f32 -> bf16 weight conversion ----------------
__global__ __launch_bounds__(256) void f32_to_bf16_kernel(
    const float* __restrict__ s, bf16_t* __restrict__ d, int n)
{
  for (int i = blockIdx.x * blockDim.x + threadIdx.x; i < n; i += gridDim.x * blockDim.x)
    d[i] = (bf16_t)s[i];
}

// ---------------- WMMA GEMM: C = (A@W + bias) * scale ----------------
// A: MxK bf16 row-major, W: KxN bf16 row-major. Block tile 128x64, 8 waves,
// each wave owns a 32x32 patch = 2x2 v_wmma_f32_16x16x32_bf16 tiles.
// A tile is staged with async copies (pure row copy); B tile is transposed
// during staging so B fragments are contiguous LDS reads.
__global__ __launch_bounds__(256) void gemm_bf16_kernel(
    const bf16_t* __restrict__ A, const bf16_t* __restrict__ W,
    const float* __restrict__ bias, float* __restrict__ outF,
    bf16_t* __restrict__ outB, int M, int K, int Nn, float scale)
{
  __shared__ bf16_t As[128][32];   // 8 KB
  __shared__ bf16_t Bt[64][32];    // 4 KB, B tile stored transposed (n-major)
  const int t    = threadIdx.x;
  const int lane = t & 31;
  const int w    = t >> 5;
  const int half = lane >> 4;      // wave32: lane groups 0-15 / 16-31
  const int lm   = lane & 15;
  const int wr   = w & 3;          // 4 row-waves
  const int wc   = w >> 2;         // 2 col-waves
  const int rowBase = blockIdx.y * 128;
  const int colBase = blockIdx.x * 64;

  const v8f zero = {};
  v8f acc[2][2];
  acc[0][0] = zero; acc[0][1] = zero; acc[1][0] = zero; acc[1][1] = zero;

  for (int k0 = 0; k0 < K; k0 += 32) {
    __syncthreads();
    {   // A tile 128x32: thread t copies half a row (16 elems = 2x16B async)
      const int r = t >> 1, hr = (t & 1) * 16;
      const bf16_t* src = A + (size_t)(rowBase + r) * K + k0 + hr;
      stage16(&As[r][hr],     src);
      stage16(&As[r][hr + 8], src + 8);
      if (k0 + 32 < K)  // speculative prefetch of the next K tile
        __builtin_prefetch(src + 32, 0, 1);
    }
    {   // B tile 32x64 -> transposed store into Bt[n][k]
      const int kk = t >> 3, nc = (t & 7) * 8;
      const v8bf wv = *(const v8bf*)(W + (size_t)(k0 + kk) * Nn + colBase + nc);
      #pragma unroll
      for (int j = 0; j < 8; ++j) Bt[nc + j][kk] = wv[j];
    }
    wait_stage();
    __syncthreads();

    #pragma unroll
    for (int mt = 0; mt < 2; ++mt) {
      Frag16 af;  // A 16x32 frag: lane lm = row M, K groups per ISA layout
      const bf16_t* ap = &As[wr * 32 + mt * 16 + lm][half * 8];
      af.h[0] = *(const v8bf*)ap;
      af.h[1] = *(const v8bf*)(ap + 16);
      #pragma unroll
      for (int nt = 0; nt < 2; ++nt) {
        Frag16 bfg; // B 32x16 frag: lane lm = col N, K groups per ISA layout (same as A)
        const bf16_t* bp = &Bt[wc * 32 + nt * 16 + lm][half * 8];
        bfg.h[0] = *(const v8bf*)bp;
        bfg.h[1] = *(const v8bf*)(bp + 16);
        acc[mt][nt] = __builtin_amdgcn_wmma_f32_16x16x32_bf16(
            false, af.v, false, bfg.v, (short)0, acc[mt][nt], false, false);
      }
    }
  }

  // epilogue: C/D layout -> lane holds rows (i + half*8), col (lm)
  #pragma unroll
  for (int mt = 0; mt < 2; ++mt) {
    #pragma unroll
    for (int nt = 0; nt < 2; ++nt) {
      const int col = colBase + wc * 32 + nt * 16 + lm;
      const float bv = bias ? bias[col] : 0.0f;
      #pragma unroll
      for (int i = 0; i < 8; ++i) {
        const int row = rowBase + wr * 32 + mt * 16 + half * 8 + i;
        const float vv = (acc[mt][nt][i] + bv) * scale;
        const size_t idx = (size_t)row * Nn + col;
        if (outF) outF[idx] = vv;
        if (outB) outB[idx] = (bf16_t)vv;
      }
    }
  }
}

// ---------------- attention diagonal ----------------
// Per (n, head e), Q/K sub-matrices are contiguous 1024x64 reshapes of 64
// activation rows.  diag[b] = exp(S[b,b]) / sum_a exp(S[a,b]); S carries the
// 1/H factor because Q,K were pre-scaled by 1/sqrt(H) in their GEMMs.
__global__ __launch_bounds__(256) void attn_diag_kernel(
    const bf16_t* __restrict__ Qb, const bf16_t* __restrict__ Kb, float* __restrict__ D)
{
  const int b0 = blockIdx.x * 64;   // 64-column slab of S
  const int e  = blockIdx.y;
  const int n  = blockIdx.z;
  const bf16_t* Qp = Qb + ((size_t)n * L_SEQ + (size_t)e * HH_DIM) * H_DIM; // 1024x64
  const bf16_t* Kp = Kb + ((size_t)n * L_SEQ + (size_t)e * HH_DIM) * H_DIM;

  __shared__ bf16_t Kt[64][64];     // K sub-rows for the 64 columns (8 KB)
  __shared__ float  colsum[64];
  __shared__ float  numer[64];

  const int t    = threadIdx.x;
  const int lane = t & 31;
  const int w    = t >> 5;
  const int half = lane >> 4;
  const int lm   = lane & 15;
  const int ar   = w & 1;           // 2 row-waves per a-chunk
  const int bc   = w >> 1;          // 4 column tiles of 16

  {   // stage K tile (64x64 bf16) via async copy: 2x16B per thread
    const int r = t >> 2, c = (t & 3) * 16;
    const bf16_t* src = Kp + (size_t)(b0 + r) * HH_DIM + c;
    stage16(&Kt[r][c],     src);
    stage16(&Kt[r][c + 8], src + 8);
  }
  if (t < 64) { colsum[t] = 0.0f; numer[t] = 0.0f; }
  wait_stage();
  __syncthreads();

  const int mycol0 = b0 + bc * 16;
  float colacc = 0.0f;

  for (int achunk = 0; achunk < L_SEQ; achunk += 32) {
    const int abase = achunk + ar * 16;
    const bf16_t* arow = Qp + (size_t)(abase + lm) * HH_DIM;
    v8f acc = {};
    #pragma unroll
    for (int ks = 0; ks < 2; ++ks) {        // K = 64 -> two 16x16x32 steps
      Frag16 af, bfg;
      af.h[0] = *(const v8bf*)(arow + ks * 32 + half * 8);
      af.h[1] = *(const v8bf*)(arow + ks * 32 + 16 + half * 8);
      const bf16_t* bp = &Kt[bc * 16 + lm][ks * 32 + half * 8];
      bfg.h[0] = *(const v8bf*)bp;
      bfg.h[1] = *(const v8bf*)(bp + 16);
      acc = __builtin_amdgcn_wmma_f32_16x16x32_bf16(
          false, af.v, false, bfg.v, (short)0, acc, false, false);
    }
    #pragma unroll
    for (int i = 0; i < 8; ++i) {           // exp + column accumulation
      const float ex = __expf(acc[i]);
      colacc += ex;
      if (abase == mycol0 && (i + half * 8) == lm)   // diagonal element
        numer[bc * 16 + lm] = ex;
    }
  }
  atomicAdd(&colsum[bc * 16 + lm], colacc);   // ds_add_f32 reduction
  __syncthreads();

  if (t < 64) {
    const int b  = b0 + t;
    const int l  = e * HH_DIM + (b >> 4);
    const int hb = b & 15;
    D[((size_t)n * L_SEQ + l) * HEADS + hb] = numer[t] / colsum[t];
  }
}

// ---------------- attnV = diag * V  (bf16 for the Wo GEMM) ----------------
__global__ __launch_bounds__(256) void scale_v_kernel(
    const float* __restrict__ D, const float* __restrict__ Vf, bf16_t* __restrict__ avb)
{
  for (size_t i = (size_t)blockIdx.x * 256 + threadIdx.x; i < NLH;
       i += (size_t)gridDim.x * 256) {
    const int h = (int)(i & (H_DIM - 1));
    const size_t nl = i >> 10;
    avb[i] = (bf16_t)(D[nl * HEADS + (h >> 6)] * Vf[i]);
  }
}

// ---------------- h = LayerNorm(attn + x) ----------------
__global__ __launch_bounds__(256) void add_ln_kernel(
    const float* __restrict__ attn, const float* __restrict__ x,
    const float* __restrict__ g, const float* __restrict__ b,
    float* __restrict__ hF, bf16_t* __restrict__ hB)
{
  const int row = blockIdx.x;
  const int t   = threadIdx.x;
  const size_t base = (size_t)row * H_DIM;
  const float4 va = ((const float4*)(attn + base))[t];
  const float4 vx = ((const float4*)(x    + base))[t];
  float v[4] = { va.x + vx.x, va.y + vx.y, va.z + vx.z, va.w + vx.w };

  float s = v[0] + v[1] + v[2] + v[3];
  float ss = v[0]*v[0] + v[1]*v[1] + v[2]*v[2] + v[3]*v[3];
  __shared__ float rs[256], rss[256];
  rs[t] = s; rss[t] = ss;
  __syncthreads();
  for (int o = 128; o > 0; o >>= 1) {
    if (t < o) { rs[t] += rs[t + o]; rss[t] += rss[t + o]; }
    __syncthreads();
  }
  const float mean = rs[0] * (1.0f / H_DIM);
  const float var  = rss[0] * (1.0f / H_DIM) - mean * mean;   // biased var
  const float rstd = rsqrtf(var + 1e-5f);
  const int c = t * 4;
  #pragma unroll
  for (int j = 0; j < 4; ++j) {
    const float hv = (v[j] - mean) * rstd * g[c + j] + b[c + j];
    hF[base + c + j] = hv;
    hB[base + c + j] = (bf16_t)hv;
  }
}

// ---------------- y = ff + h  (new x, f32 + bf16) ----------------
__global__ __launch_bounds__(256) void add_store_kernel(
    const float* __restrict__ a, const float* __restrict__ b,
    float* __restrict__ oF, bf16_t* __restrict__ oB)
{
  for (size_t i = (size_t)blockIdx.x * 256 + threadIdx.x; i < NLH;
       i += (size_t)gridDim.x * 256) {
    const float v = a[i] + b[i];
    oF[i] = v; oB[i] = (bf16_t)v;
  }
}

__global__ __launch_bounds__(256) void copy_f32_kernel(
    const float* __restrict__ s, float* __restrict__ d, size_t n)
{
  for (size_t i = (size_t)blockIdx.x * 256 + threadIdx.x; i < n;
       i += (size_t)gridDim.x * 256) d[i] = s[i];
}

// ---------------- orchestration ----------------
extern "C" void kernel_launch(void* const* d_in, const int* in_sizes, int n_in,
                              void* d_out, int out_size, void* d_ws, size_t ws_size,
                              hipStream_t stream) {
  (void)in_sizes; (void)n_in; (void)ws_size;
  const int*   X    = (const int*)  d_in[0];
  const float* tok  = (const float*)d_in[1];
  const float* pos  = (const float*)d_in[2];
  const float* Wq   = (const float*)d_in[3];
  const float* bq   = (const float*)d_in[4];
  const float* Wk   = (const float*)d_in[5];
  const float* bk   = (const float*)d_in[6];
  const float* Wv   = (const float*)d_in[7];
  const float* bv   = (const float*)d_in[8];
  const float* Wo   = (const float*)d_in[9];
  const float* bo   = (const float*)d_in[10];
  const float* Wff  = (const float*)d_in[11];
  const float* bff  = (const float*)d_in[12];
  const float* g1   = (const float*)d_in[13];
  const float* b1   = (const float*)d_in[14];
  float* out = (float*)d_out;

  // workspace carve-up (256B aligned)
  char* ws = (char*)d_ws;
  size_t off = 0;
  auto alloc = [&](size_t bytes) -> void* {
    void* p = ws + off;
    off = (off + bytes + 255) & ~(size_t)255;
    return p;
  };
  float*  xF    = (float*) alloc(NLH * 4);
  bf16_t* xB    = (bf16_t*)alloc(NLH * 2);
  bf16_t* qB    = (bf16_t*)alloc(NLH * 2);
  bf16_t* kB    = (bf16_t*)alloc(NLH * 2);
  float*  vF    = (float*) alloc(NLH * 4);
  bf16_t* avB   = (bf16_t*)alloc(NLH * 2);
  float*  tmpF  = (float*) alloc(NLH * 4);   // attn_out, then ff output
  float*  hF    = (float*) alloc(NLH * 4);
  bf16_t* hB    = (bf16_t*)alloc(NLH * 2);
  float*  Dbuf  = (float*) alloc((size_t)NL * HEADS * 4);
  bf16_t* wBf   = (bf16_t*)alloc(5 * EH * 2);
  bf16_t* wqB = wBf, *wkB = wBf + EH, *wvB = wBf + 2*EH, *woB = wBf + 3*EH, *wffB = wBf + 4*EH;

  const dim3 gemmGrid(H_DIM / 64, NL / 128);   // 16 x 16 blocks
  const dim3 attnGrid(L_SEQ / 64, HEADS, N_B); // 16 x 16 x 2
  const int  ewBlocks = 2048;                  // grid-stride elementwise

  embed_kernel<<<NL, 256, 0, stream>>>(X, tok, pos, xF, xB);

  for (int d = 0; d < DEPTH; ++d) {
    const size_t wo_ = (size_t)d * EH;
    const size_t bo_ = (size_t)d * H_DIM;
    f32_to_bf16_kernel<<<512, 256, 0, stream>>>(Wq  + wo_, wqB,  (int)EH);
    f32_to_bf16_kernel<<<512, 256, 0, stream>>>(Wk  + wo_, wkB,  (int)EH);
    f32_to_bf16_kernel<<<512, 256, 0, stream>>>(Wv  + wo_, wvB,  (int)EH);
    f32_to_bf16_kernel<<<512, 256, 0, stream>>>(Wo  + wo_, woB,  (int)EH);
    f32_to_bf16_kernel<<<512, 256, 0, stream>>>(Wff + wo_, wffB, (int)EH);

    // Q,K scaled by 1/sqrt(H); V kept f32 for diag*V
    gemm_bf16_kernel<<<gemmGrid, 256, 0, stream>>>(xB, wqB, bq + bo_, nullptr, qB,
                                                   NL, E_DIM, H_DIM, INV_SQRT_H);
    gemm_bf16_kernel<<<gemmGrid, 256, 0, stream>>>(xB, wkB, bk + bo_, nullptr, kB,
                                                   NL, E_DIM, H_DIM, INV_SQRT_H);
    gemm_bf16_kernel<<<gemmGrid, 256, 0, stream>>>(xB, wvB, bv + bo_, vF, nullptr,
                                                   NL, E_DIM, H_DIM, 1.0f);

    attn_diag_kernel<<<attnGrid, 256, 0, stream>>>(qB, kB, Dbuf);
    scale_v_kernel<<<ewBlocks, 256, 0, stream>>>(Dbuf, vF, avB);

    gemm_bf16_kernel<<<gemmGrid, 256, 0, stream>>>(avB, woB, bo + bo_, tmpF, nullptr,
                                                   NL, H_DIM, H_DIM, 1.0f);
    add_ln_kernel<<<NL, 256, 0, stream>>>(tmpF, xF, g1 + bo_, b1 + bo_, hF, hB);

    gemm_bf16_kernel<<<gemmGrid, 256, 0, stream>>>(hB, wffB, bff + bo_, tmpF, nullptr,
                                                   NL, H_DIM, H_DIM, 1.0f);
    add_store_kernel<<<ewBlocks, 256, 0, stream>>>(tmpF, hF, xF, xB);
  }

  copy_f32_kernel<<<ewBlocks, 256, 0, stream>>>(xF, out, (size_t)out_size);
}
